// CMix_x060moe_86887188398515
// MI455X (gfx1250) — matrix operand order repaired and checked by reading, not compiled, not run on hardware
//
#include <hip/hip_runtime.h>

typedef unsigned short u16;
typedef __attribute__((ext_vector_type(16))) __bf16 bf16x16;
typedef __attribute__((ext_vector_type(8)))  float  f32x8;
typedef __attribute__((ext_vector_type(4)))  int    i32x4;

#define B_   8
#define T_   2048
#define C_   1024
#define DFF_ 4096
#define DFFE_ 2048
#define E_   8
#define CAP_ 2048
#define NT_  (B_ * T_)
#define HASHP 5099u

#define BM 128
#define BN 128
#define BK 32

#define AS1 __attribute__((address_space(1)))
#define AS3 __attribute__((address_space(3)))

typedef AS1 i32x4* gvec128_t;   // global-AS pointer to 16B
typedef AS3 i32x4* lvec128_t;   // LDS-AS pointer to 16B

#if __has_builtin(__builtin_amdgcn_global_load_async_to_lds_b128) && \
    __has_builtin(__builtin_amdgcn_s_wait_asynccnt)
#define USE_ASYNC_LDS 1
#else
#define USE_ASYNC_LDS 0
#endif

__device__ __forceinline__ u16 f32_to_bf16(float f) {
  unsigned u = __float_as_uint(f);
  u += 0x7FFFu + ((u >> 16) & 1u);   // round-to-nearest-even
  return (u16)(u >> 16);
}

// ---------------------------------------------------------------------------
// 1) Token shift: xk/xr = x + (xprev - x) * maa, emitted as bf16
// ---------------------------------------------------------------------------
__global__ __launch_bounds__(256) void prep_kernel(
    const float* __restrict__ x, const float* __restrict__ shift,
    const float* __restrict__ maak, const float* __restrict__ maar,
    u16* __restrict__ xk, u16* __restrict__ xr)
{
  int row = blockIdx.x;              // 0..NT-1  (= b*T + t)
  int b   = row >> 11;               // T = 2048
  int t   = row & (T_ - 1);
  int c   = threadIdx.x * 4;
  size_t off = (size_t)row * C_ + c;

  float4 xv = *(const float4*)(x + off);
  float4 pv = (t == 0) ? *(const float4*)(shift + (size_t)b * C_ + c)
                       : *(const float4*)(x + off - C_);
  float4 mk = *(const float4*)(maak + c);
  float4 mr = *(const float4*)(maar + c);

  float dx0 = pv.x - xv.x, dx1 = pv.y - xv.y, dx2 = pv.z - xv.z, dx3 = pv.w - xv.w;

  unsigned k01 = (unsigned)f32_to_bf16(xv.x + dx0 * mk.x) |
                 ((unsigned)f32_to_bf16(xv.y + dx1 * mk.y) << 16);
  unsigned k23 = (unsigned)f32_to_bf16(xv.z + dx2 * mk.z) |
                 ((unsigned)f32_to_bf16(xv.w + dx3 * mk.w) << 16);
  unsigned r01 = (unsigned)f32_to_bf16(xv.x + dx0 * mr.x) |
                 ((unsigned)f32_to_bf16(xv.y + dx1 * mr.y) << 16);
  unsigned r23 = (unsigned)f32_to_bf16(xv.z + dx2 * mr.z) |
                 ((unsigned)f32_to_bf16(xv.w + dx3 * mr.w) << 16);

  *(uint2*)(xk + off) = make_uint2(k01, k23);
  *(uint2*)(xr + off) = make_uint2(r01, r23);
}

// ---------------------------------------------------------------------------
// 2) Weight convert + transpose: f32 [K,N] row-major -> bf16 [N,K]
// ---------------------------------------------------------------------------
__global__ __launch_bounds__(256) void transpose_kernel(
    const float* __restrict__ src, u16* __restrict__ dst, int K, int N)
{
  size_t bOff = (size_t)blockIdx.z * K * N;
  int idx = blockIdx.x * 256 + threadIdx.x;      // over dst elements
  int n = idx / K, k = idx - n * K;
  dst[bOff + idx] = f32_to_bf16(src[bOff + (size_t)k * N + n]);
}

// ---------------------------------------------------------------------------
// 3) Hash routing: expert id + first-come position within expert queue.
// ---------------------------------------------------------------------------
__global__ void route_kernel(const int* __restrict__ tok,
                             int* __restrict__ eidx, int* __restrict__ pos)
{
  int lane = threadIdx.x;            // 32 lanes
  int base[E_];
#pragma unroll
  for (int e = 0; e < E_; ++e) base[e] = 0;

  for (int c0 = 0; c0 < NT_; c0 += 32) {
    unsigned t = (unsigned)tok[c0 + lane];
    unsigned e = (t * HASHP) & 7u;
    int myPos = 0;
#pragma unroll
    for (int ex = 0; ex < E_; ++ex) {
      unsigned bal = __builtin_amdgcn_ballot_w32(e == (unsigned)ex);
      if (e == (unsigned)ex)
        myPos = base[ex] + __builtin_popcount(bal & ((1u << lane) - 1u));
      base[ex] += __builtin_popcount(bal);
    }
    eidx[c0 + lane] = (int)e;
    pos[c0 + lane]  = myPos;
  }
}

// ---------------------------------------------------------------------------
// 4) Dispatch scatter: buf[e][p][*] = xk[i][*] for kept tokens
// ---------------------------------------------------------------------------
__global__ __launch_bounds__(256) void scatter_kernel(
    const u16* __restrict__ xk, const int* __restrict__ eidx,
    const int* __restrict__ pos, u16* __restrict__ buf)
{
  int i = blockIdx.x;
  int p = pos[i];
  if (p >= CAP_) return;
  int e = eidx[i];
  const uint2* s = (const uint2*)(xk + (size_t)i * C_);
  uint2* d = (uint2*)(buf + ((size_t)e * CAP_ + p) * C_);
  d[threadIdx.x] = s[threadIdx.x];   // 256 * 8B = 1024 bf16
}

// ---------------------------------------------------------------------------
// 5) Tiled bf16 WMMA GEMM:  C[M,N] = A[M,K] @ B[K,N],  B given as BT[N,K].
//    Block tile 128x128x32, double-buffered LDS, 8 waves; each wave owns a
//    32x64 (2x4 WMMA) tile -> 8 v_wmma per K-step.
//    Global->LDS staging uses CDNA5 async-to-LDS (ASYNCcnt) when available.
//    EPI 0: f32 out.  EPI 1: bf16( relu(x)^2 ) out (feeds next GEMM).
// ---------------------------------------------------------------------------
template<int EPI>
__global__ __launch_bounds__(256) void wmma_gemm_kernel(
    const u16* __restrict__ A, const u16* __restrict__ BT,
    void* __restrict__ Cout, int M, int N, int K,
    long long aBatch, long long bBatch, long long cBatch)
{
  __shared__ u16 As[2][BM * BK];
  __shared__ u16 Bs[2][BN * BK];

  const int tid  = threadIdx.x;
  const int lane = tid & 31;
  const int wv   = tid >> 5;
  const int wm   = wv >> 1;          // 0..3 (M direction, 32 rows each)
  const int wn   = wv & 1;           // 0..1 (N direction, 64 cols each)
  const int laneHi = lane >> 4;      // 0/1
  const int lane15 = lane & 15;

  const size_t aBase = (size_t)blockIdx.z * aBatch;
  const size_t bBase = (size_t)blockIdx.z * bBatch;
  const size_t cBase = (size_t)blockIdx.z * cBatch;

  const int m0 = blockIdx.y * BM;
  const int n0 = blockIdx.x * BN;

  const int srow  = tid >> 1;        // 0..127 (staging row, A and B tiles)
  const int shalf = tid & 1;         // which 16-elem half of the 32-wide row

  f32x8 acc[2][4] = {};

  // ---- staging: copy one 128x32 A tile + 128x32 BT tile into LDS buffer sb
  auto stage = [&](int sb, int k0) {
    const u16* ga = A  + aBase + (size_t)(m0 + srow) * K + k0 + shalf * 16;
    const u16* gb = BT + bBase + (size_t)(n0 + srow) * K + k0 + shalf * 16;
    u16* la = &As[sb][srow * BK + shalf * 16];
    u16* lb = &Bs[sb][srow * BK + shalf * 16];
#if USE_ASYNC_LDS
    __builtin_amdgcn_global_load_async_to_lds_b128(
        (gvec128_t)(unsigned long long)ga,
        (lvec128_t)(AS3 void*)la, 0, 0);
    __builtin_amdgcn_global_load_async_to_lds_b128(
        (gvec128_t)(unsigned long long)(ga + 8),
        (lvec128_t)(AS3 void*)(la + 8), 0, 0);
    __builtin_amdgcn_global_load_async_to_lds_b128(
        (gvec128_t)(unsigned long long)gb,
        (lvec128_t)(AS3 void*)lb, 0, 0);
    __builtin_amdgcn_global_load_async_to_lds_b128(
        (gvec128_t)(unsigned long long)(gb + 8),
        (lvec128_t)(AS3 void*)(lb + 8), 0, 0);
#else
    uint4 a0 = ((const uint4*)ga)[0], a1 = ((const uint4*)ga)[1];
    uint4 b0 = ((const uint4*)gb)[0], b1 = ((const uint4*)gb)[1];
    *(uint4*)la = a0; *(uint4*)(la + 8) = a1;
    *(uint4*)lb = b0; *(uint4*)(lb + 8) = b1;
    if (k0 + BK < K)
      __builtin_prefetch(ga + BK, 0, 1);
#endif
  };

  // ---- compute: 2 A-frags, 4 B-frags, 8 WMMA from LDS buffer sb
  auto compute = [&](int sb) {
    // A fragment (16-bit A 16x32): lanes0-15 K[0..7]+[16..23],
    // lanes16-31 K[8..15]+[24..31]; row = lane&15.
    bf16x16 af[2];
#pragma unroll
    for (int mi = 0; mi < 2; ++mi) {
      int row = wm * 32 + mi * 16 + lane15;
      const uint4* p = (const uint4*)&As[sb][row * BK + laneHi * 8];
      union { uint4 u[2]; bf16x16 v; } fr;
      fr.u[0] = p[0];
      fr.u[1] = p[2];                // +32 bytes (K+16)
      af[mi] = fr.v;
    }
    // B fragment (32x16): lanes0-15 K[0..7]+[16..23], lanes16-31 K[8..15]+[24..31]; col = lane&15.
    bf16x16 bfr[4];
#pragma unroll
    for (int ni = 0; ni < 4; ++ni) {
      int col = wn * 64 + ni * 16 + lane15;
      const uint4* p = (const uint4*)&Bs[sb][col * BK + laneHi * 8];
      union { uint4 u[2]; bf16x16 v; } fr;
      fr.u[0] = p[0];
      fr.u[1] = p[2];
      bfr[ni] = fr.v;
    }
#pragma unroll
    for (int mi = 0; mi < 2; ++mi)
#pragma unroll
      for (int ni = 0; ni < 4; ++ni)
        acc[mi][ni] = __builtin_amdgcn_wmma_f32_16x16x32_bf16(
            false, af[mi], false, bfr[ni], (short)0, acc[mi][ni], false, false);
  };

  // ---- software-pipelined main loop (double-buffered LDS)
  stage(0, 0);
#if USE_ASYNC_LDS
  __builtin_amdgcn_s_wait_asynccnt(0);
#endif
  __syncthreads();

  int s = 0;
  for (int k0 = BK; k0 < K; k0 += BK) {
    stage(s ^ 1, k0);
    compute(s);
#if USE_ASYNC_LDS
    __builtin_amdgcn_s_wait_asynccnt(0);
#endif
    __syncthreads();
    s ^= 1;
  }
  compute(s);

  // ---- epilogue. C/D: lanes0-15 col n rows M=j; lanes16-31 rows M=8+j.
#pragma unroll
  for (int mi = 0; mi < 2; ++mi) {
#pragma unroll
    for (int ni = 0; ni < 4; ++ni) {
      int n    = n0 + wn * 64 + ni * 16 + lane15;
      int mrow = m0 + wm * 32 + mi * 16 + laneHi * 8;
#pragma unroll
      for (int j = 0; j < 8; ++j) {
        float v = acc[mi][ni][j];
        size_t off = cBase + (size_t)(mrow + j) * N + n;
        if (EPI == 0) {
          ((float*)Cout)[off] = v;
        } else {
          float rv = v > 0.f ? v : 0.f;
          rv *= rv;
          ((u16*)Cout)[off] = f32_to_bf16(rv);
        }
      }
    }
  }
}

// ---------------------------------------------------------------------------
// 6) Combine: out = sigmoid(r) * (kv + moe_gather)
// ---------------------------------------------------------------------------
__global__ __launch_bounds__(256) void combine_kernel(
    const float* __restrict__ r, const float* __restrict__ kv,
    const float* __restrict__ o, const int* __restrict__ eidx,
    const int* __restrict__ pos, float* __restrict__ out)
{
  int i = blockIdx.x;
  int c = threadIdx.x * 4;
  size_t off = (size_t)i * C_ + c;
  int p = pos[i], e = eidx[i];

  float4 kvv = *(const float4*)(kv + off);
  float4 ov = make_float4(0.f, 0.f, 0.f, 0.f);
  if (p < CAP_)
    ov = *(const float4*)(o + ((size_t)e * CAP_ + p) * C_ + c);
  float4 rv = *(const float4*)(r + off);

  float4 res;
  res.x = (kvv.x + ov.x) / (1.f + __expf(-rv.x));
  res.y = (kvv.y + ov.y) / (1.f + __expf(-rv.y));
  res.z = (kvv.z + ov.z) / (1.f + __expf(-rv.z));
  res.w = (kvv.w + ov.w) / (1.f + __expf(-rv.w));
  *(float4*)(out + off) = res;
}

// ---------------------------------------------------------------------------
extern "C" void kernel_launch(void* const* d_in, const int* in_sizes, int n_in,
                              void* d_out, int out_size, void* d_ws, size_t ws_size,
                              hipStream_t stream)
{
  const float* x     = (const float*)d_in[0];
  const float* shift = (const float*)d_in[1];
  const int*   tok   = (const int*)d_in[2];
  const float* maak  = (const float*)d_in[3];
  const float* maar  = (const float*)d_in[4];
  const float* Wk    = (const float*)d_in[5];
  const float* Wv    = (const float*)d_in[6];
  const float* Wr    = (const float*)d_in[7];
  const float* Wek   = (const float*)d_in[8];
  const float* Wev   = (const float*)d_in[9];
  float* out = (float*)d_out;

  // workspace carve-up (256B aligned)
  char* w = (char*)d_ws;
  auto alloc = [&](size_t bytes) -> char* {
    char* p = w; w += (bytes + 255) & ~(size_t)255; return p;
  };
  u16*   xk   = (u16*)  alloc((size_t)NT_ * C_ * 2);
  u16*   xr   = (u16*)  alloc((size_t)NT_ * C_ * 2);
  u16*   WkT  = (u16*)  alloc((size_t)DFF_ * C_ * 2);
  u16*   WvT  = (u16*)  alloc((size_t)C_ * DFF_ * 2);
  u16*   WrT  = (u16*)  alloc((size_t)C_ * C_ * 2);
  u16*   WekT = (u16*)  alloc((size_t)E_ * DFFE_ * C_ * 2);
  u16*   WevT = (u16*)  alloc((size_t)E_ * C_ * DFFE_ * 2);
  u16*   ksq  = (u16*)  alloc((size_t)NT_ * DFF_ * 2);
  float* kv   = (float*)alloc((size_t)NT_ * C_ * 4);
  u16*   buf  = (u16*)  alloc((size_t)E_ * CAP_ * C_ * 2);
  u16*   hbuf = (u16*)  alloc((size_t)E_ * CAP_ * DFFE_ * 2);
  float* obuf = (float*)alloc((size_t)E_ * CAP_ * C_ * 4);
  float* rlog = (float*)alloc((size_t)NT_ * C_ * 4);
  int*   eidx = (int*)  alloc((size_t)NT_ * 4);
  int*   pos  = (int*)  alloc((size_t)NT_ * 4);

  (void)hipMemsetAsync(buf, 0, (size_t)E_ * CAP_ * C_ * 2, stream);

  prep_kernel<<<NT_, 256, 0, stream>>>(x, shift, maak, maar, xk, xr);

  transpose_kernel<<<dim3((C_ * DFF_) / 256, 1, 1), 256, 0, stream>>>(Wk, WkT, C_, DFF_);
  transpose_kernel<<<dim3((DFF_ * C_) / 256, 1, 1), 256, 0, stream>>>(Wv, WvT, DFF_, C_);
  transpose_kernel<<<dim3((C_ * C_) / 256, 1, 1), 256, 0, stream>>>(Wr, WrT, C_, C_);
  transpose_kernel<<<dim3((C_ * DFFE_) / 256, 1, E_), 256, 0, stream>>>(Wek, WekT, C_, DFFE_);
  transpose_kernel<<<dim3((DFFE_ * C_) / 256, 1, E_), 256, 0, stream>>>(Wev, WevT, DFFE_, C_);

  route_kernel<<<1, 32, 0, stream>>>(tok, eidx, pos);
  scatter_kernel<<<NT_, 256, 0, stream>>>(xk, eidx, pos, buf);

  // dense FFN: ksq = sqrelu(xk @ Wk)  [16384 x 4096]
  wmma_gemm_kernel<1><<<dim3(DFF_ / BN, NT_ / BM, 1), 256, 0, stream>>>(
      xk, WkT, ksq, NT_, DFF_, C_, 0, 0, 0);
  // kv = ksq @ Wv  [16384 x 1024] f32
  wmma_gemm_kernel<0><<<dim3(C_ / BN, NT_ / BM, 1), 256, 0, stream>>>(
      ksq, WvT, kv, NT_, C_, DFF_, 0, 0, 0);
  // MoE: h = sqrelu(buf @ Wek)  per expert
  wmma_gemm_kernel<1><<<dim3(DFFE_ / BN, CAP_ / BM, E_), 256, 0, stream>>>(
      buf, WekT, hbuf, CAP_, DFFE_, C_,
      (long long)CAP_ * C_, (long long)DFFE_ * C_, (long long)CAP_ * DFFE_);
  // o = h @ Wev  per expert, f32
  wmma_gemm_kernel<0><<<dim3(C_ / BN, CAP_ / BM, E_), 256, 0, stream>>>(
      hbuf, WevT, obuf, CAP_, C_, DFFE_,
      (long long)CAP_ * DFFE_, (long long)C_ * DFFE_, (long long)CAP_ * C_);
  // gate logits: r = xr @ Wr  f32
  wmma_gemm_kernel<0><<<dim3(C_ / BN, NT_ / BM, 1), 256, 0, stream>>>(
      xr, WrT, rlog, NT_, C_, C_, 0, 0, 0);

  combine_kernel<<<NT_, 256, 0, stream>>>(rlog, kv, obuf, eidx, pos, out);
}
